// GraphFormation_52269751992747
// MI455X (gfx1250) — compile-verified, matrix operand order checked by reading, not run on hardware
//
#include <hip/hip_runtime.h>

// ---------------------------------------------------------------------------
// MI455X (gfx1250) implementation of the KAN graph-formation pipeline.
//
// Roofline: x is 67MB fp32 -> ~2.9us at 23.3TB/s HBM; total matrix work is
// ~5.5 GFLOP (32768 rows/branch through a KAN: K=64 base GEMM + K=512 spline
// GEMM, Nout=64). Compute is tiny vs the WMMA ceiling, so the design:
//  - read x exactly once with float4 loads (two max-pool kernels),
//  - run all KAN GEMMs as v_wmma_f32_16x16x32_f16 (f32 accumulate); operands
//    are bounded (tanh in (-1,1), spline bases in [0,1], clipped coeffs),
//  - fuse prod/LN/tanh/silu/B-spline A-operand construction in registers,
//  - store f16 weights K-group-major so each WMMA B half-operand is a single
//    aligned ds_load_b128 (no per-element packing),
//  - branch-free tanh/sigmoid via v_exp_f32 + raw v_rcp_f32
//    (__builtin_amdgcn_rcpf) -- no EXEC churn and no IEEE
//    div_scale/div_fmas/div_fixup expansions (which __frcp_rn produces).
// Wave32: 128-thread blocks = 4 waves, each wave owns a 16-row M-tile.
// ---------------------------------------------------------------------------

typedef __attribute__((ext_vector_type(16))) _Float16 v16h;
typedef __attribute__((ext_vector_type(8)))  _Float16 v8h;
typedef __attribute__((ext_vector_type(8)))  float    v8f;

#define GAL_SMEM  (16384 + 16384 + 65536 + 8192 + 1024)   // hn, hln, scB(f16), bW(f16), misc
#define KAN_SMEM  (16384 + 65536 + 8192)                  // X, scB(f16), bW(f16)

// Raw hardware reciprocal: single v_rcp_f32 (~1 ulp) -- plenty for these
// activations. (__frcp_rn would expand to the full IEEE division sequence.)
__device__ __forceinline__ float rcp_fast(float x)  { return __builtin_amdgcn_rcpf(x); }
__device__ __forceinline__ float sigmoidf_(float x) { return rcp_fast(1.0f + __expf(-x)); }
__device__ __forceinline__ float siluf_(float x)    { return x * sigmoidf_(x); }

// Branch-free tanh: v_max + v_min + v_exp_f32 + v_rcp_f32, no EXEC changes.
// Inputs here are bounded (layernorm outputs / KAN accumulators), clamp is safe.
__device__ __forceinline__ float tanhf_fast(float x) {
  float xc = fminf(fmaxf(x, -15.f), 15.f);
  float e = __expf(2.f * xc);
  return (e - 1.f) * rcp_fast(e + 1.f);
}

// Cubic B-spline (Cox-de Boor) on the uniform extended grid g[i] = -2.2+0.4i,
// 12 knots -> 8 cubic basis values. Fully unrolled -> stays in VGPRs.
__device__ __forceinline__ void bspline8(float x, float f[8]) {
  const float g0 = -2.2f, h = 0.4f;
  float b[11];
#pragma unroll
  for (int i = 0; i < 11; ++i) {
    float gi = g0 + h * (float)i;
    b[i] = (x >= gi && x < gi + h) ? 1.0f : 0.0f;
  }
#pragma unroll
  for (int k = 1; k <= 3; ++k) {
    float inv = 1.0f / (h * (float)k);   // compile-time constant
#pragma unroll
    for (int i = 0; i < 11 - k; ++i) {
      float gi = g0 + h * (float)i;
      b[i] = (x - gi) * inv * b[i] + ((gi + h * (float)(k + 1)) - x) * inv * b[i + 1];
    }
  }
#pragma unroll
  for (int i = 0; i < 8; ++i) f[i] = b[i];
}

#define SHUF16(a, b) __builtin_shufflevector((a), (b), 0, 1, 2, 3, 4, 5, 6, 7, \
                                             8, 9, 10, 11, 12, 13, 14, 15)

// One wave computes a 16-row M-tile x 64 output cols of the KAN:
//   out = silu(t) @ baseW  +  bspline8(t) @ scB     (t = per-dim input, tval(d))
// Weights live in LDS K-group-major: w8[(k>>3)*512 + col*8 + (k&7)] so the two
// 8-consecutive-k halves of a WMMA B operand are two aligned b128 LDS loads.
template <typename F>
__device__ __forceinline__ void kan_wave(v8f acc[4], int hi, int nl,
                                         const v8h* __restrict__ sbW8,
                                         const v8h* __restrict__ sscB8,
                                         const F& tval) {
  // base path: K = 64 -> 2 K-steps of 32
#pragma unroll
  for (int kk = 0; kk < 2; ++kk) {
    v8h a0, a1;
#pragma unroll
    for (int p = 0; p < 8; ++p) a0[p] = (_Float16)siluf_(tval(kk * 32 + hi * 8 + p));
#pragma unroll
    for (int p = 0; p < 8; ++p) a1[p] = (_Float16)siluf_(tval(kk * 32 + 16 + hi * 8 + p));
    v16h A = SHUF16(a0, a1);
    const int g0 = kk * 4 + hi, g1 = kk * 4 + 2 + hi;
#pragma unroll
    for (int nt = 0; nt < 4; ++nt) {
      const int col = nt * 16 + nl;
      v16h Bv = SHUF16(sbW8[g0 * 64 + col], sbW8[g1 * 64 + col]);
      acc[nt] = __builtin_amdgcn_wmma_f32_16x16x32_f16(false, A, false, Bv,
                                                       (short)0, acc[nt], false, false);
    }
  }
  // spline path: K = 512 -> 16 K-steps of 32. Each lane needs exactly two
  // dims' basis values per K-step: i = kk*4+hi and kk*4+2+hi.
#pragma unroll 2
  for (int kk = 0; kk < 16; ++kk) {
    float f0[8], f1[8];
    bspline8(tval(kk * 4 + hi), f0);
    bspline8(tval(kk * 4 + 2 + hi), f1);
    v8h a0, a1;
#pragma unroll
    for (int p = 0; p < 8; ++p) { a0[p] = (_Float16)f0[p]; a1[p] = (_Float16)f1[p]; }
    v16h A = SHUF16(a0, a1);
    const int g0 = kk * 4 + hi, g1 = kk * 4 + 2 + hi;
#pragma unroll
    for (int nt = 0; nt < 4; ++nt) {
      const int col = nt * 16 + nl;
      v16h Bv = SHUF16(sscB8[g0 * 64 + col], sscB8[g1 * 64 + col]);
      acc[nt] = __builtin_amdgcn_wmma_f32_16x16x32_f16(false, A, false, Bv,
                                                       (short)0, acc[nt], false, false);
    }
  }
}

// ---------------------------------------------------------------------------
// Weight prep: sc[o][i][j] = clip(coeffs,-5,5)*spline_w[o][i] -> f16 in
// K-group-major layout: dst[(k>>3)*512 + o*8 + (k&7)], k = i*8+j.
// base_w[o][k] -> same layout (8 groups).
__global__ void prep_kan_kernel(const float* __restrict__ base_w,
                                const float* __restrict__ spline_w,
                                const float* __restrict__ coeffs,
                                _Float16* __restrict__ scB,
                                _Float16* __restrict__ bWh) {
  int idx = blockIdx.x * blockDim.x + threadIdx.x;  // 32768
  int o = idx & 63, k = idx >> 6;
  int i = k >> 3, j = k & 7;
  float c = coeffs[(o * 64 + i) * 8 + j];
  c = fminf(fmaxf(c, -5.f), 5.f);
  scB[(k >> 3) * 512 + o * 8 + (k & 7)] = (_Float16)(c * spline_w[o * 64 + i]);
  if (idx < 4096) {
    int oo = idx & 63, kk = idx >> 6;
    bWh[(kk >> 3) * 512 + oo * 8 + (kk & 7)] = (_Float16)(base_w[oo * 64 + kk]);
  }
}

__global__ void wm_kernel(const float* __restrict__ W, float* __restrict__ wmo) {
  int d = threadIdx.x;  // 64
  float s = 0.f;
  for (int k = 0; k < 64; ++k) s += W[d * 64 + k];
  wmo[d] = s * (1.f / 64.f);
}

// temporal pool: tfeat[b,nt,c] = max_{q<64} |x[b,c,nt*64+q]|
__global__ void tpool_kernel(const float* __restrict__ x, float* __restrict__ tfeat) {
  int idx = blockIdx.x * blockDim.x + threadIdx.x;  // 262144
  int b = idx >> 15, rem = idx & 32767, c = rem >> 6, nt = rem & 63;
  const float4* p = (const float4*)(x + (size_t)(b * 512 + c) * 4096 + nt * 64);
  float mx = 0.f;
#pragma unroll 4
  for (int q = 0; q < 16; ++q) {
    float4 v = p[q];
    mx = fmaxf(mx, fmaxf(fmaxf(fabsf(v.x), fabsf(v.y)), fmaxf(fabsf(v.z), fabsf(v.w))));
  }
  tfeat[(size_t)(b * 64 + nt) * 512 + c] = mx;
}

// spatial pool: sfeat[b,ns,sp] = max over 8 channels x 128 samples of |x|
__global__ void spool_kernel(const float* __restrict__ x, float* __restrict__ sfeat) {
  int gw = blockIdx.x * 8 + (threadIdx.x >> 5);  // 16384 waves
  int lane = threadIdx.x & 31;
  int b = gw >> 11, rem = gw & 2047, ns = rem >> 5, sp = rem & 31;
  int cc = lane & 7, ch = lane >> 3;
  const float4* p = (const float4*)(x + (size_t)(b * 512 + ns * 8 + cc) * 4096 + sp * 128 + ch * 32);
  float mx = 0.f;
#pragma unroll
  for (int q = 0; q < 8; ++q) {
    float4 v = p[q];
    mx = fmaxf(mx, fmaxf(fmaxf(fabsf(v.x), fabsf(v.y)), fmaxf(fabsf(v.z), fabsf(v.w))));
  }
  for (int msk = 1; msk < 32; msk <<= 1) mx = fmaxf(mx, __shfl_xor(mx, msk, 32));
  if (lane == 0) sfeat[(size_t)(b * 64 + ns) * 32 + sp] = mx;
}

// projection + sinusoidal PE: out[row,d] = feat[row,:]@W[d,:] + bias[d] + pe(n,d)
__global__ void proj_kernel(const float* __restrict__ feat, const float* __restrict__ W,
                            const float* __restrict__ bias, float* __restrict__ out, int K) {
  int idx = blockIdx.x * blockDim.x + threadIdx.x;  // 32768
  int row = idx >> 6, d = idx & 63, n = row & 63;
  const float* fr = feat + (size_t)row * K;
  const float* wr = W + (size_t)d * K;
  float s = bias[d];
  for (int c = 0; c < K; c += 4) {
    float4 a = *(const float4*)(fr + c);
    float4 b = *(const float4*)(wr + c);
    s += a.x * b.x + a.y * b.y + a.z * b.z + a.w * b.w;
  }
  int i2 = d >> 1;
  float freq = __expf(-(float)(2 * i2) * (9.2103403719761836f / 64.f));
  float ang = (float)n * freq;
  s += (d & 1) ? __cosf(ang) : __sinf(ang);
  out[idx] = s;
}

// LayerNorm(D=64) + L2-normalize, one wave per row.
__global__ __launch_bounds__(32) void lnnorm_kernel(const float* __restrict__ gin,
                                                    const float* __restrict__ g,
                                                    const float* __restrict__ bia,
                                                    float* __restrict__ hln,
                                                    float* __restrict__ hn) {
  int row = blockIdx.x, lane = threadIdx.x;
  float a0 = gin[row * 64 + lane], a1 = gin[row * 64 + 32 + lane];
  float s = a0 + a1;
  for (int m = 1; m < 32; m <<= 1) s += __shfl_xor(s, m, 32);
  float mean = s * (1.f / 64.f);
  float d0 = a0 - mean, d1 = a1 - mean;
  float v = d0 * d0 + d1 * d1;
  for (int m = 1; m < 32; m <<= 1) v += __shfl_xor(v, m, 32);
  float rstd = rsqrtf(v * (1.f / 64.f) + 1e-5f);
  float h0 = d0 * rstd * g[lane] + bia[lane];
  float h1 = d1 * rstd * g[lane + 32] + bia[lane + 32];
  float nn = h0 * h0 + h1 * h1;
  for (int m = 1; m < 32; m <<= 1) nn += __shfl_xor(nn, m, 32);
  float inv = rcp_fast(fmaxf(sqrtf(nn), 1e-12f));
  hln[row * 64 + lane] = h0;        hln[row * 64 + 32 + lane] = h1;
  hn [row * 64 + lane] = h0 * inv;  hn [row * 64 + 32 + lane] = h1 * inv;
}

// ---------------------------------------------------------------------------
// Core fused kernel: one block per (b, n). Computes, for all 64 m:
//   prod = hn[n]*hn[m] -> LN -> t=tanh -> KAN (WMMA) -> ko=tanh -> score=ko.Wm
// then softmax over m and h_att[n] = attn @ h_ln.
__global__ __launch_bounds__(128) void gal_kernel(const float* __restrict__ hn_g,
                                                  const float* __restrict__ hln_g,
                                                  const _Float16* __restrict__ scB_g,
                                                  const _Float16* __restrict__ bW_g,
                                                  const float* __restrict__ lnpg,
                                                  const float* __restrict__ lnpb,
                                                  const float* __restrict__ wm_g,
                                                  float* __restrict__ hatt_g) {
  extern __shared__ char smem[];
  float*    s_hn  = (float*)smem;                  // 4096 f
  float*    s_hln = s_hn + 4096;                   // 4096 f
  _Float16* s_scB = (_Float16*)(s_hln + 4096);     // 32768 h
  _Float16* s_bW  = s_scB + 32768;                 // 4096 h
  float*    s_g   = (float*)(s_bW + 4096);
  float*    s_b   = s_g + 64;
  float*    s_wm  = s_b + 64;
  float*    s_sc  = s_wm + 64;

  const int tid = threadIdx.x;
  const int b = blockIdx.x >> 6, n = blockIdx.x & 63;

  { const float4* src = (const float4*)(hn_g + (size_t)b * 4096);
    float4* dst = (float4*)s_hn;
    for (int i = tid; i < 1024; i += 128) dst[i] = src[i]; }
  { const float4* src = (const float4*)(hln_g + (size_t)b * 4096);
    float4* dst = (float4*)s_hln;
    for (int i = tid; i < 1024; i += 128) dst[i] = src[i]; }
  { const uint4* src = (const uint4*)scB_g; uint4* dst = (uint4*)s_scB;
    for (int i = tid; i < 4096; i += 128) dst[i] = src[i]; }
  { const uint4* src = (const uint4*)bW_g; uint4* dst = (uint4*)s_bW;
    for (int i = tid; i < 512; i += 128) dst[i] = src[i]; }
  if (tid < 64) { s_g[tid] = lnpg[tid]; s_b[tid] = lnpb[tid]; s_wm[tid] = wm_g[tid]; }
  __syncthreads();

  const int lane = tid & 31, w = tid >> 5;
  const int nl = lane & 15, hi = lane >> 4;
  const int m = w * 16 + nl;                 // A-matrix row for this lane
  const float* hn_n = s_hn + n * 64;
  const float* hn_m = s_hn + m * 64;

  float sum = 0.f, sq = 0.f;
#pragma unroll 8
  for (int d = 0; d < 64; ++d) { float pr = hn_n[d] * hn_m[d]; sum += pr; sq += pr * pr; }
  const float mean = sum * (1.f / 64.f);
  const float rstd = rsqrtf(sq * (1.f / 64.f) - mean * mean + 1e-5f);

  auto tval = [&](int d) -> float {
    float pr = hn_n[d] * hn_m[d];
    return tanhf_fast((pr - mean) * rstd * s_g[d] + s_b[d]);
  };

  v8f acc[4];
#pragma unroll
  for (int nt = 0; nt < 4; ++nt)
#pragma unroll
    for (int r = 0; r < 8; ++r) acc[nt][r] = 0.f;

  kan_wave(acc, hi, nl, (const v8h*)s_bW, (const v8h*)s_scB, tval);

  // scores[m] = sum_o tanh(ko[m][o]) * Wm[o]; D layout: row = r + hi*8
#pragma unroll
  for (int r = 0; r < 8; ++r) {
    float s = 0.f;
#pragma unroll
    for (int nt = 0; nt < 4; ++nt) s += tanhf_fast(acc[nt][r]) * s_wm[nt * 16 + nl];
    s += __shfl_xor(s, 1, 32); s += __shfl_xor(s, 2, 32);
    s += __shfl_xor(s, 4, 32); s += __shfl_xor(s, 8, 32);
    if (nl == 0) s_sc[w * 16 + r + hi * 8] = s;
  }
  __syncthreads();

  if (tid < 64) {  // softmax(scores/8) and h_att = attn @ h_ln ; tid = output dim
    float mx = -3.4e38f;
#pragma unroll 8
    for (int mm = 0; mm < 64; ++mm) mx = fmaxf(mx, s_sc[mm]);
    float den = 0.f, accum = 0.f;
    for (int mm = 0; mm < 64; ++mm) {
      float e = __expf((s_sc[mm] - mx) * 0.125f);
      den += e;
      accum += e * s_hln[mm * 64 + tid];
    }
    hatt_g[((size_t)(b * 64 + n)) * 64 + tid] = accum * rcp_fast(den);
  }
}

// Generic KAN over 64 rows per block (kan_proj / kan_direct).
// mode 0: Y = kan(X); mode 2: Y = (Y + kan(X)) * bn_g/sqrt(1+1e-5) + bn_b
__global__ __launch_bounds__(128) void kan_rows_kernel(const float* __restrict__ X,
                                                       const _Float16* __restrict__ scB_g,
                                                       const _Float16* __restrict__ bW_g,
                                                       float* __restrict__ Y,
                                                       const float* __restrict__ post_g,
                                                       const float* __restrict__ post_b,
                                                       int mode) {
  extern __shared__ char smem[];
  float*    s_X   = (float*)smem;                 // 4096 f
  _Float16* s_scB = (_Float16*)(s_X + 4096);      // 32768 h
  _Float16* s_bW  = s_scB + 32768;                // 4096 h

  const int tid = threadIdx.x;
  const int row0 = blockIdx.x * 64;

  { const float4* src = (const float4*)(X + (size_t)row0 * 64);
    float4* dst = (float4*)s_X;
    for (int i = tid; i < 1024; i += 128) dst[i] = src[i]; }
  { const uint4* src = (const uint4*)scB_g; uint4* dst = (uint4*)s_scB;
    for (int i = tid; i < 4096; i += 128) dst[i] = src[i]; }
  { const uint4* src = (const uint4*)bW_g; uint4* dst = (uint4*)s_bW;
    for (int i = tid; i < 512; i += 128) dst[i] = src[i]; }
  __syncthreads();

  const int lane = tid & 31, w = tid >> 5;
  const int nl = lane & 15, hi = lane >> 4;
  const float* xr = s_X + (w * 16 + nl) * 64;
  auto tval = [&](int d) -> float { return tanhf_fast(xr[d]); };

  v8f acc[4];
#pragma unroll
  for (int nt = 0; nt < 4; ++nt)
#pragma unroll
    for (int r = 0; r < 8; ++r) acc[nt][r] = 0.f;

  kan_wave(acc, hi, nl, (const v8h*)s_bW, (const v8h*)s_scB, tval);

#pragma unroll
  for (int r = 0; r < 8; ++r) {
#pragma unroll
    for (int nt = 0; nt < 4; ++nt) {
      const int col = nt * 16 + nl;
      const size_t gi = (size_t)(row0 + w * 16 + r + hi * 8) * 64 + col;
      float v = acc[nt][r];
      if (mode == 0) Y[gi] = v;
      else { float t = Y[gi] + v; Y[gi] = t * (post_g[col] * 0.9999950000374996f) + post_b[col]; }
    }
  }
}

// KAN score head (out_f = 1) + sigmoid; one wave per row.
__global__ __launch_bounds__(32) void kscore_kernel(const float* __restrict__ X,
                                                    const float* __restrict__ bw,
                                                    const float* __restrict__ spw,
                                                    const float* __restrict__ cf,
                                                    float* __restrict__ sc) {
  int row = blockIdx.x, lane = threadIdx.x;
  float s = 0.f;
#pragma unroll
  for (int q = 0; q < 2; ++q) {
    int d = lane + q * 32;
    float t = tanhf_fast(X[row * 64 + d]);
    float c = siluf_(t) * bw[d];
    float f[8]; bspline8(t, f);
    float sw = spw[d];
#pragma unroll
    for (int j = 0; j < 8; ++j) {
      float cc = fminf(fmaxf(cf[d * 8 + j], -5.f), 5.f);
      c += f[j] * cc * sw;
    }
    s += c;
  }
  for (int m = 1; m < 32; m <<= 1) s += __shfl_xor(s, m, 32);
  if (lane == 0) sc[row] = sigmoidf_(s);
}

// Stable top-k(32) of 64 via rank counting (matches lax.top_k tie-break),
// gather gated rows hg = X*score into descending-score order.
__global__ __launch_bounds__(64) void pool_kernel(const float* __restrict__ X,
                                                  const float* __restrict__ sc,
                                                  float* __restrict__ out) {
  __shared__ float s_s[64];
  int b = blockIdx.x, i = threadIdx.x;
  s_s[i] = sc[b * 64 + i];
  __syncthreads();
  float si = s_s[i];
  int rank = 0;
  for (int j = 0; j < 64; ++j) {
    float sj = s_s[j];
    rank += (sj > si) || (sj == si && j < i);
  }
  if (rank < 32) {
    const float* src = X + (size_t)(b * 64 + i) * 64;
    float* dst = out + (size_t)(b * 32 + rank) * 64;
    for (int d = 0; d < 64; ++d) dst[d] = src[d] * si;
  }
}

// ---------------------------------------------------------------------------
extern "C" void kernel_launch(void* const* d_in, const int* in_sizes, int n_in,
                              void* d_out, int out_size, void* d_ws, size_t ws_size,
                              hipStream_t stream) {
  (void)in_sizes; (void)n_in; (void)out_size; (void)ws_size;
  const float* x       = (const float*)d_in[0];
  const float* tproj_w = (const float*)d_in[1];
  const float* tproj_b = (const float*)d_in[2];
  const float* sproj_w = (const float*)d_in[3];
  const float* sproj_b = (const float*)d_in[4];
  auto F = [&](int i) { return (const float*)d_in[i]; };

  // workspace carve (~2.5 MB total)
  char* p = (char*)d_ws;
  auto alloc = [&](size_t bytes) { char* r = p; p += (bytes + 255) & ~(size_t)255; return r; };
  float* tfeat = (float*)alloc(262144 * sizeof(float));
  float* sfeat = (float*)alloc(16384 * sizeof(float));
  float *gin[2], *hlnb[2], *hnb[2], *hatt[2], *yb[2], *psc[2], *wmv[2];
  _Float16 *scB[2][3], *bW[2][3];
  for (int br = 0; br < 2; ++br) {
    gin[br]  = (float*)alloc(32768 * 4);
    hlnb[br] = (float*)alloc(32768 * 4);
    hnb[br]  = (float*)alloc(32768 * 4);
    hatt[br] = (float*)alloc(32768 * 4);
    yb[br]   = (float*)alloc(32768 * 4);
    psc[br]  = (float*)alloc(512 * 4);
    wmv[br]  = (float*)alloc(64 * 4);
    for (int k = 0; k < 3; ++k) {
      scB[br][k] = (_Float16*)alloc(32768 * 2);
      bW[br][k]  = (_Float16*)alloc(4096 * 2);
    }
  }

  // allow >64KB dynamic LDS on the WMMA kernels (WGP has 320KB)
  (void)hipFuncSetAttribute((const void*)gal_kernel,
                            hipFuncAttributeMaxDynamicSharedMemorySize, GAL_SMEM);
  (void)hipFuncSetAttribute((const void*)kan_rows_kernel,
                            hipFuncAttributeMaxDynamicSharedMemorySize, KAN_SMEM);

  // weight prep (f16 K-group-major B operands) + W_att mean
  for (int br = 0; br < 2; ++br) {
    int gb = br == 0 ? 5 : 21;
    const int koffsets[3] = {0, 4, 7};  // kan_att, kan_proj, kan_direct
    for (int k = 0; k < 3; ++k) {
      int base = gb + koffsets[k];
      prep_kan_kernel<<<128, 256, 0, stream>>>(F(base), F(base + 1), F(base + 2),
                                               scB[br][k], bW[br][k]);
    }
    wm_kernel<<<1, 64, 0, stream>>>(F(gb + 3), wmv[br]);
  }

  // stream x once: pooled node features
  tpool_kernel<<<1024, 256, 0, stream>>>(x, tfeat);
  spool_kernel<<<2048, 256, 0, stream>>>(x, sfeat);

  // projections + positional encoding
  proj_kernel<<<128, 256, 0, stream>>>(tfeat, tproj_w, tproj_b, gin[0], 512);
  proj_kernel<<<128, 256, 0, stream>>>(sfeat, sproj_w, sproj_b, gin[1], 32);

  float* outf = (float*)d_out;
  for (int br = 0; br < 2; ++br) {
    int gb = br == 0 ? 5 : 21;
    int pb = br == 0 ? 37 : 40;
    lnnorm_kernel<<<512, 32, 0, stream>>>(gin[br], F(gb + 10), F(gb + 11),
                                          hlnb[br], hnb[br]);
    gal_kernel<<<512, 128, GAL_SMEM, stream>>>(hnb[br], hlnb[br], scB[br][0], bW[br][0],
                                               F(gb + 12), F(gb + 13), wmv[br], hatt[br]);
    kan_rows_kernel<<<8, 128, KAN_SMEM, stream>>>(hatt[br], scB[br][1], bW[br][1],
                                                  yb[br], nullptr, nullptr, 0);
    kan_rows_kernel<<<8, 128, KAN_SMEM, stream>>>(hlnb[br], scB[br][2], bW[br][2],
                                                  yb[br], F(gb + 14), F(gb + 15), 2);
    kscore_kernel<<<512, 32, 0, stream>>>(yb[br], F(pb), F(pb + 1), F(pb + 2), psc[br]);
    pool_kernel<<<8, 64, 0, stream>>>(yb[br], psc[br], outf + br * 16384);
  }
}
